// HGT_63221918597829
// MI455X (gfx1250) — matrix operand order repaired and checked by reading, not compiled, not run on hardware
//
#include <hip/hip_runtime.h>
#include <hip/hip_bf16.h>
#include <math.h>

#define HID   256
#define NH    8
#define DH    32
#define N1C   60
#define N2C   240
#define N3C   300
#define LTOT  600
#define LPAD  608
#define BB    32
#define SS    512
#define DBERT 768
#define KOUT  (LTOT * HID)   /* 153600 */

typedef __attribute__((ext_vector_type(16))) _Float16 v16h;
typedef __attribute__((ext_vector_type(8)))  _Float16 v8h;
typedef __attribute__((ext_vector_type(8)))  float    v8f;

#define NEG_INF (-__builtin_inff())

// ================= WMMA fragment helpers (NT GEMM) =========================
// A 16x32 f16 fragment: lane (m + 16h) holds K in {8h..8h+7} U {16+8h..16+8h+7}
// B 32x16 f16 fragment: lane (n + 16h) holds K in {8h..8h+7} U {16+8h..16+8h+7}

__device__ __forceinline__ v16h frag_a_f32(const float* base, long ld, int rowMax,
                                           int r0, int k0) {
  int lane = threadIdx.x & 31;
  int r = r0 + (lane & 15);
  if (r > rowMax) r = rowMax;               // clamp: keep loads in-bounds
  int h = lane >> 4;
  const float* p = base + (long)r * ld + k0 + h * 8;
  v16h a;
#pragma unroll
  for (int j = 0; j < 8; ++j) a[j] = (_Float16)p[j];
#pragma unroll
  for (int j = 0; j < 8; ++j) a[8 + j] = (_Float16)p[16 + j];
  return a;
}

__device__ __forceinline__ v16h frag_b_f32(const float* baseT, long ld, int colMax,
                                           int n0, int k0) {
  int lane = threadIdx.x & 31;
  int n = n0 + (lane & 15);
  if (n > colMax) n = colMax;
  int h = lane >> 4;
  const float* p = baseT + (long)n * ld + k0 + h * 8;
  v16h b;
#pragma unroll
  for (int j = 0; j < 8; ++j) b[j] = (_Float16)p[j];
#pragma unroll
  for (int j = 0; j < 8; ++j) b[8 + j] = (_Float16)p[16 + j];
  return b;
}

// f16 fragments: pure 16B vector loads, zero conversion VALU.
__device__ __forceinline__ v16h frag_a_f16(const _Float16* base, long ld, int rowMax,
                                           int r0, int k0) {
  int lane = threadIdx.x & 31;
  int r = r0 + (lane & 15);
  if (r > rowMax) r = rowMax;
  int h = lane >> 4;
  const _Float16* p = base + (long)r * ld + k0 + h * 8;
  v8h lo = *(const v8h*)p;            // K = 8h..8h+7
  v8h hi = *(const v8h*)(p + 16);     // K = 16+8h..16+8h+7
  return __builtin_shufflevector(lo, hi, 0, 1, 2, 3, 4, 5, 6, 7,
                                 8, 9, 10, 11, 12, 13, 14, 15);
}

__device__ __forceinline__ v16h frag_b_f16(const _Float16* baseT, long ld, int colMax,
                                           int n0, int k0) {
  int lane = threadIdx.x & 31;
  int n = n0 + (lane & 15);
  if (n > colMax) n = colMax;
  int h = lane >> 4;
  const _Float16* p = baseT + (long)n * ld + k0 + h * 8;
  v8h lo = *(const v8h*)p;
  v8h hi = *(const v8h*)(p + 16);
  return __builtin_shufflevector(lo, hi, 0, 1, 2, 3, 4, 5, 6, 7,
                                 8, 9, 10, 11, 12, 13, 14, 15);
}

__device__ __forceinline__ v8f wmma_f16(v16h a, v16h b, v8f c) {
  return __builtin_amdgcn_wmma_f32_16x16x32_f16(false, a, false, b, (short)0, c,
                                                false, false);
}

__device__ __forceinline__ void store_f16(_Float16* C, long ldc, int m0, int n0,
                                          int Mmax, v8f c) {
  int lane = threadIdx.x & 31;
  int n  = n0 + (lane & 15);
  int mb = m0 + ((lane >> 4) << 3);
#pragma unroll
  for (int r = 0; r < 8; ++r) {
    int m = mb + r;
    if (m < Mmax) C[(long)m * ldc + n] = (_Float16)c[r];
  }
}

// ================= elementwise f32 -> f16 pre-convert ======================
__global__ void k_cvt_f32_f16(const float* __restrict__ src, _Float16* __restrict__ dst,
                              int n) {
  int i = blockIdx.x * 256 + threadIdx.x;
  if (i < n) dst[i] = (_Float16)src[i];
}

// ================= Stage 0: text_enc = tanh(text_hidden @ Wt^T + bt) =======
// 8 accumulators per wave: one A fragment (50MB text_hidden stream) feeds 8
// WMMAs -> A re-read 2x instead of 16x. Writes f16 te + transposed teT.
__global__ void k_text_enc(const float* __restrict__ th, const _Float16* __restrict__ Wt16,
                           const float* __restrict__ bt, _Float16* __restrict__ te16,
                           _Float16* __restrict__ teT16) {
  int gid   = blockIdx.x * 8 + (threadIdx.x >> 5);  // 2048 waves
  int mtile = gid >> 1;                             // 1024 M tiles
  int ng    = gid & 1;                              // ntile group: 0..7 / 8..15
  int m0 = mtile * 16;
  v8f c[8] = {};
  for (int k = 0; k < DBERT; k += 32) {
    v16h a = frag_a_f32(th, DBERT, BB * SS - 1, m0, k);
#pragma unroll
    for (int j = 0; j < 8; ++j) {
      v16h b = frag_b_f16(Wt16, DBERT, HID - 1, (ng * 8 + j) * 16, k);
      c[j] = wmma_f16(a, b, c[j]);
    }
  }
  int lane = threadIdx.x & 31;
  int mb = m0 + ((lane >> 4) << 3);
#pragma unroll
  for (int j = 0; j < 8; ++j) {
    int n = (ng * 8 + j) * 16 + (lane & 15);
    float bias = bt[n];
#pragma unroll
    for (int r = 0; r < 8; ++r) {
      int m = mb + r;
      _Float16 v = (_Float16)tanhf(c[j][r] + bias);
      te16[(long)m * HID + n] = v;
      int bI = m >> 9, s = m & (SS - 1);
      teT16[((long)bI * HID + n) * SS + s] = v;
    }
  }
}

// ================= Stage 1: le = label_feat @ Wl^T + bl (f32) ==============
__global__ void k_le0(const float* __restrict__ lf, const float* __restrict__ Wl,
                      const float* __restrict__ bl, float* __restrict__ le) {
  int mtile = blockIdx.y, ntile = blockIdx.x;
  int m0 = mtile * 16, n0 = ntile * 16;
  v8f c = {};
  for (int k = 0; k < DBERT; k += 32) {
    v16h a = frag_a_f32(lf, DBERT, LTOT - 1, m0, k);
    v16h b = frag_b_f32(Wl, DBERT, HID - 1, n0, k);
    c = wmma_f16(a, b, c);
  }
  int lane = threadIdx.x & 31;
  int n  = n0 + (lane & 15);
  int mb = m0 + ((lane >> 4) << 3);
  float bias = bl[n];
#pragma unroll
  for (int r = 0; r < 8; ++r) {
    int m = mb + r;
    if (m < LTOT) le[(long)m * HID + n] = c[r] + bias;
  }
}

// ================= per-relation-type 256x256 projection ====================
__global__ void k_proj3(const float* __restrict__ X, const float* __restrict__ W3,
                        float* __restrict__ Out, const float* __restrict__ leAdd) {
  int t = blockIdx.z;
  int rowOff = (t == 0) ? 0 : (t == 1) ? N1C : (N1C + N2C);
  int Mblk   = (t == 0) ? N1C : (t == 1) ? N2C : N3C;
  int mtile  = blockIdx.y;
  if (mtile * 16 >= Mblk) return;                 // uniform per block
  int m0 = rowOff + mtile * 16, n0 = blockIdx.x * 16;
  int rMax = rowOff + Mblk - 1;
  const float* W = W3 + (long)t * HID * HID;
  v8f c = {};
  for (int k = 0; k < HID; k += 32) {
    v16h a = frag_a_f32(X, HID, rMax, m0, k);
    v16h b = frag_b_f32(W, HID, HID - 1, n0, k);
    c = wmma_f16(a, b, c);
  }
  int lane = threadIdx.x & 31;
  int n  = n0 + (lane & 15);
  int mb = m0 + ((lane >> 4) << 3);
#pragma unroll
  for (int r = 0; r < 8; ++r) {
    int m = mb + r;
    if (m <= rMax) {
      long idx = (long)m * HID + n;
      Out[idx] = leAdd ? (tanhf(c[r]) + leAdd[idx]) : c[r];
    }
  }
}

// ================= PAQ[l, h*32+e] = sum_d Q[l,h*32+d] * PA[e,d] ============
__global__ void k_paq(const float* __restrict__ Q, const float* __restrict__ PA,
                      float* __restrict__ PAQ) {
  int h = blockIdx.z, mtile = blockIdx.y, ntile = blockIdx.x;
  int m0 = mtile * 16;
  v16h a = frag_a_f32(Q + h * DH, HID, LTOT - 1, m0, 0);
  v16h b = frag_b_f32(PA, DH, DH - 1, ntile * 16, 0);
  v8f c = {};
  c = wmma_f16(a, b, c);
  int lane = threadIdx.x & 31;
  int n  = h * DH + ntile * 16 + (lane & 15);
  int mb = m0 + ((lane >> 4) << 3);
#pragma unroll
  for (int r = 0; r < 8; ++r) {
    int m = mb + r;
    if (m < LTOT) PAQ[(long)m * HID + n] = c[r];
  }
}

// ================= PMVT[h][e][l] = sum_d PM[e,d] * V[l,h*32+d] =============
__global__ void k_pmvt(const float* __restrict__ PM, const float* __restrict__ V,
                       float* __restrict__ PMVT) {
  int h = blockIdx.z, mtile = blockIdx.y, ntile = blockIdx.x;
  int m0 = mtile * 16, n0 = ntile * 16;
  v16h a = frag_a_f32(PM, DH, DH - 1, m0, 0);
  v16h b = frag_b_f32(V + h * DH, HID, LTOT - 1, n0, 0);
  v8f c = {};
  c = wmma_f16(a, b, c);
  float* out = PMVT + (long)h * DH * LPAD;
  int lane = threadIdx.x & 31;
  int n  = n0 + (lane & 15);
  int mb = m0 + ((lane >> 4) << 3);
#pragma unroll
  for (int r = 0; r < 8; ++r) {
    int e = mb + r;
    out[(long)e * LPAD + n] = (n < LTOT) ? c[r] : 0.f;   // zero K-pad
  }
}

// ================= Sc[h][q][k] = sum_e PAQ[q,h*32+e] * K[k,h*32+e] =========
__global__ void k_sc(const float* __restrict__ PAQ, const float* __restrict__ Kb,
                     float* __restrict__ Sc) {
  int h = blockIdx.z, mtile = blockIdx.y, ntile = blockIdx.x;
  v16h a = frag_a_f32(PAQ + h * DH, HID, LTOT - 1, mtile * 16, 0);
  v16h b = frag_b_f32(Kb + h * DH, HID, LTOT - 1, ntile * 16, 0);
  v8f c = {};
  c = wmma_f16(a, b, c);
  float* out = Sc + (long)h * LPAD * LPAD;
  int lane = threadIdx.x & 31;
  int n  = ntile * 16 + (lane & 15);
  int mb = mtile * 16 + ((lane >> 4) << 3);
#pragma unroll
  for (int r = 0; r < 8; ++r) out[(long)(mb + r) * LPAD + n] = c[r];
}

// ================= masked softmax over k (zero-block + att mask) ===========
__global__ void k_softmax_att(float* __restrict__ Att, const int* __restrict__ attRaw) {
  int q = blockIdx.x, h = blockIdx.y;
  float* row = Att + ((long)h * LPAD + q) * LPAD;
  const float scale = 0.17677669529663687f;   // 1/sqrt(32)
  bool q1 = q < N1C, q3 = q >= (N1C + N2C);
  __shared__ float sred[256];
  int t = threadIdx.x;
  float vals[3]; int ks[3]; int cnt = 0;
  float mx = NEG_INF;
  for (int k = t; k < LTOT; k += 256) {
    bool k1 = k < N1C, k3 = k >= (N1C + N2C);
    bool Z = (q1 && k3) || (q3 && k1);
    float v = Z ? 0.f : row[k] * scale;               // Sc = where(Z,0,Sc)
    if (attRaw[(long)q * LTOT + k] == 0) v = NEG_INF; // + additive mask
    vals[cnt] = v; ks[cnt] = k; ++cnt;
    mx = fmaxf(mx, v);
  }
  sred[t] = mx; __syncthreads();
  for (int o = 128; o > 0; o >>= 1) { if (t < o) sred[t] = fmaxf(sred[t], sred[t + o]); __syncthreads(); }
  float rmax = sred[0]; __syncthreads();
  float se = 0.f;
  for (int i = 0; i < cnt; ++i) { float e = expf(vals[i] - rmax); vals[i] = e; se += e; }
  sred[t] = se; __syncthreads();
  for (int o = 128; o > 0; o >>= 1) { if (t < o) sred[t] += sred[t + o]; __syncthreads(); }
  float inv = 1.f / sred[0];
  for (int i = 0; i < cnt; ++i) {
    int k = ks[i];
    bool k1 = k < N1C, k3 = k >= (N1C + N2C);
    bool Z = (q1 && k3) || (q3 && k1);
    row[k] = Z ? 0.f : vals[i] * inv;                 // Watt *= (1 - Z)
  }
  if (t < LPAD - LTOT) row[LTOT + t] = 0.f;           // zero K-pad
}

// ================= M[q, h*32+e] = sum_k Watt[h][q][k] * PMVT[h][e][k] ======
__global__ void k_msum(const float* __restrict__ Att, const float* __restrict__ PMVT,
                       float* __restrict__ Mb) {
  int h = blockIdx.z, mtile = blockIdx.y, ntile = blockIdx.x;
  const float* A  = Att + (long)h * LPAD * LPAD;
  const float* Bt = PMVT + (long)h * DH * LPAD;
  v8f c = {};
  for (int k = 0; k < LPAD; k += 32) {
    v16h a = frag_a_f32(A, LPAD, LTOT - 1, mtile * 16, k);
    v16h b = frag_b_f32(Bt, LPAD, DH - 1, ntile * 16, k);
    c = wmma_f16(a, b, c);
  }
  int lane = threadIdx.x & 31;
  int n  = h * DH + ntile * 16 + (lane & 15);
  int mb = mtile * 16 + ((lane >> 4) << 3);
#pragma unroll
  for (int r = 0; r < 8; ++r) {
    int m = mb + r;
    if (m < LTOT) Mb[(long)m * HID + n] = c[r];
  }
}

// ================= aw[b][l][s] = sum_e le[l,e] * te[b,s,e]  (2x2 tiled) ====
__global__ void k_aw(const _Float16* __restrict__ le16, const _Float16* __restrict__ te16,
                     _Float16* __restrict__ aw16) {
  int ntp = blockIdx.x * 8 + (threadIdx.x >> 5);  // 16 ntile-pairs
  int mtp = blockIdx.y, b = blockIdx.z;           // 19 mtile-pairs, 32 batches
  int m0 = mtp * 32, n0 = ntp * 32;
  const _Float16* B = te16 + (long)b * SS * HID;
  v8f c00 = {}, c01 = {}, c10 = {}, c11 = {};
  for (int k = 0; k < HID; k += 32) {
    v16h a0 = frag_a_f16(le16, HID, LTOT - 1, m0, k);
    v16h a1 = frag_a_f16(le16, HID, LTOT - 1, m0 + 16, k);
    v16h b0 = frag_b_f16(B, HID, SS - 1, n0, k);
    v16h b1 = frag_b_f16(B, HID, SS - 1, n0 + 16, k);
    c00 = wmma_f16(a0, b0, c00); c01 = wmma_f16(a0, b1, c01);
    c10 = wmma_f16(a1, b0, c10); c11 = wmma_f16(a1, b1, c11);
  }
  _Float16* out = aw16 + (long)b * LTOT * SS;
  store_f16(out, SS, m0,      n0,      LTOT, c00);
  store_f16(out, SS, m0,      n0 + 16, LTOT, c01);
  store_f16(out, SS, m0 + 16, n0,      LTOT, c10);
  store_f16(out, SS, m0 + 16, n0 + 16, LTOT, c11);
}

// ================= pad-masked softmax over s (f16 in/out, f32 math) ========
__global__ void k_softmax_aw(_Float16* __restrict__ aw16, const int* __restrict__ toks) {
  int l = blockIdx.x, b = blockIdx.y;
  _Float16* row = aw16 + ((long)b * LTOT + l) * SS;
  const int* tk = toks + (long)b * SS;
  __shared__ float sred[256];
  int t = threadIdx.x;
  int w0 = tk[t], w1 = tk[t + 256];
  float v0 = (float)row[t]       + ((w0 == 0 || w0 == 101 || w0 == 102) ? NEG_INF : 0.f);
  float v1 = (float)row[t + 256] + ((w1 == 0 || w1 == 101 || w1 == 102) ? NEG_INF : 0.f);
  sred[t] = fmaxf(v0, v1); __syncthreads();
  for (int o = 128; o > 0; o >>= 1) { if (t < o) sred[t] = fmaxf(sred[t], sred[t + o]); __syncthreads(); }
  float rmax = sred[0]; __syncthreads();
  float e0 = expf(v0 - rmax), e1 = expf(v1 - rmax);
  sred[t] = e0 + e1; __syncthreads();
  for (int o = 128; o > 0; o >>= 1) { if (t < o) sred[t] += sred[t + o]; __syncthreads(); }
  float inv = 1.f / sred[0];
  row[t]       = (_Float16)(e0 * inv);
  row[t + 256] = (_Float16)(e1 * inv);
}

// ================= feat[b][l][e] = sum_s aw[b][l][s] * teT[b][e][s] (2x2) ==
__global__ void k_feat(const _Float16* __restrict__ aw16, const _Float16* __restrict__ teT16,
                       _Float16* __restrict__ feat16) {
  int ntp = blockIdx.x * 8 + (threadIdx.x >> 5);  // 8 ntile-pairs (HID/32)
  int mtp = blockIdx.y, b = blockIdx.z;           // 19 mtile-pairs
  int m0 = mtp * 32, n0 = ntp * 32;
  const _Float16* A  = aw16 + (long)b * LTOT * SS;
  const _Float16* Bt = teT16 + (long)b * HID * SS;
  v8f c00 = {}, c01 = {}, c10 = {}, c11 = {};
  for (int k = 0; k < SS; k += 32) {
    v16h a0 = frag_a_f16(A, SS, LTOT - 1, m0, k);
    v16h a1 = frag_a_f16(A, SS, LTOT - 1, m0 + 16, k);
    v16h b0 = frag_b_f16(Bt, SS, HID - 1, n0, k);
    v16h b1 = frag_b_f16(Bt, SS, HID - 1, n0 + 16, k);
    c00 = wmma_f16(a0, b0, c00); c01 = wmma_f16(a0, b1, c01);
    c10 = wmma_f16(a1, b0, c10); c11 = wmma_f16(a1, b1, c11);
  }
  _Float16* out = feat16 + (long)b * KOUT;
  store_f16(out, HID, m0,      n0,      LTOT, c00);
  store_f16(out, HID, m0,      n0 + 16, LTOT, c01);
  store_f16(out, HID, m0 + 16, n0,      LTOT, c10);
  store_f16(out, HID, m0 + 16, n0 + 16, LTOT, c11);
}

// ================= output head: split-K over Wout (368 MB stream) ==========
// Both M tiles (B=32) fused per wave: one Wout fragment feeds 2 WMMAs,
// so the dominant stream is read exactly once.
__global__ void k_final(const _Float16* __restrict__ feat16, const float* __restrict__ Wout,
                        float* __restrict__ partial) {
  int gid   = blockIdx.x * 8 + (threadIdx.x >> 5);  // 1216 waves
  int ks    = gid / 38;                             // 32 K-splits
  int ntile = gid % 38;                             // 38 N tiles (L=600)
  const int KCH = KOUT / 32;                        // 4800 per split
  int k0 = ks * KCH;
  const float* wrow = Wout + (long)(ntile * 16) * KOUT;
  v8f c0 = {}, c1 = {};
  for (int k = k0; k < k0 + KCH; k += 32) {
    __builtin_prefetch(wrow + k + 512, 0, 0);       // stream-ahead on Wout
    v16h b  = frag_b_f32(Wout, KOUT, LTOT - 1, ntile * 16, k);
    v16h a0 = frag_a_f16(feat16, KOUT, BB - 1, 0, k);
    v16h a1 = frag_a_f16(feat16, KOUT, BB - 1, 16, k);
    c0 = wmma_f16(a0, b, c0);
    c1 = wmma_f16(a1, b, c1);
  }
  float* out = partial + (long)ks * BB * LPAD;
  int lane = threadIdx.x & 31;
  int n  = ntile * 16 + (lane & 15);
  int mb = (lane >> 4) << 3;
#pragma unroll
  for (int r = 0; r < 8; ++r) {
    out[(long)(mb + r) * LPAD + n]      = c0[r];
    out[(long)(mb + 16 + r) * LPAD + n] = c1[r];
  }
}

// ================= deterministic split-K reduce + bias + sigmoid ===========
__global__ void k_reduce(const float* __restrict__ partial, const float* __restrict__ bout,
                         float* __restrict__ out) {
  int i = blockIdx.x * 256 + threadIdx.x;
  if (i >= BB * LTOT) return;
  int b = i / LTOT, l = i % LTOT;
  float s = bout[l];
#pragma unroll 4
  for (int ks = 0; ks < 32; ++ks) s += partial[(long)ks * BB * LPAD + (long)b * LPAD + l];
  out[i] = 1.f / (1.f + expf(-s));
}

extern "C" void kernel_launch(void* const* d_in, const int* in_sizes, int n_in,
                              void* d_out, int out_size, void* d_ws, size_t ws_size,
                              hipStream_t stream) {
  const int*   toks   = (const int*)d_in[0];
  const float* th     = (const float*)d_in[1];
  const float* lf     = (const float*)d_in[2];
  const int*   attRaw = (const int*)d_in[3];
  const float* Wt     = (const float*)d_in[4];
  const float* bt     = (const float*)d_in[5];
  const float* Wl     = (const float*)d_in[6];
  const float* bl     = (const float*)d_in[7];
  const float* WK     = (const float*)d_in[8];
  const float* WQ     = (const float*)d_in[9];
  const float* WV     = (const float*)d_in[10];
  const float* PA     = (const float*)d_in[11];
  const float* PM     = (const float*)d_in[12];
  const float* WA     = (const float*)d_in[13];
  const float* Wout   = (const float*)d_in[14];
  const float* bout   = (const float*)d_in[15];
  float* out = (float*)d_out;

  // ---- f32 workspace region ----
  float* w    = (float*)d_ws;
  float* le   = w;  w += (long)LPAD * HID;
  float* Kb   = w;  w += (long)LPAD * HID;
  float* Qb   = w;  w += (long)LPAD * HID;
  float* Vb   = w;  w += (long)LPAD * HID;
  float* PAQ  = w;  w += (long)LPAD * HID;
  float* PMVT = w;  w += (long)NH * DH * LPAD;
  float* Att  = w;  w += (long)NH * LPAD * LPAD;   // 11.8 MB
  float* Mb   = w;  w += (long)LPAD * HID;
  float* part = w;  w += (long)32 * BB * LPAD;     // 2.5 MB
  // ---- f16 workspace region (all offsets 16B-aligned) ----
  _Float16* hws   = (_Float16*)w;
  _Float16* te16  = hws;  hws += (long)BB * SS * HID;   // 8.4 MB
  _Float16* teT16 = hws;  hws += (long)BB * HID * SS;   // 8.4 MB
  _Float16* aw16  = hws;  hws += (long)BB * LTOT * SS;  // 19.7 MB
  _Float16* ft16  = hws;  hws += (long)BB * KOUT;       // 9.8 MB
  _Float16* le16  = hws;  hws += (long)LTOT * HID;
  _Float16* Wt16  = hws;  hws += (long)HID * DBERT;     // total ~66 MB

  k_cvt_f32_f16<<<(HID * DBERT + 255) / 256, 256, 0, stream>>>(Wt, Wt16, HID * DBERT);
  k_text_enc<<<256, 256, 0, stream>>>(th, Wt16, bt, te16, teT16);
  k_le0<<<dim3(16, 38), 32, 0, stream>>>(lf, Wl, bl, le);
  for (int layer = 0; layer < 2; ++layer) {
    k_proj3<<<dim3(16, 19, 3), 32, 0, stream>>>(le, WK, Kb, nullptr);
    k_proj3<<<dim3(16, 19, 3), 32, 0, stream>>>(le, WQ, Qb, nullptr);
    k_proj3<<<dim3(16, 19, 3), 32, 0, stream>>>(le, WV, Vb, nullptr);
    k_paq<<<dim3(2, 38, NH), 32, 0, stream>>>(Qb, PA, PAQ);
    k_pmvt<<<dim3(38, 2, NH), 32, 0, stream>>>(PM, Vb, PMVT);
    k_sc<<<dim3(38, 38, NH), 32, 0, stream>>>(PAQ, Kb, Att);
    k_softmax_att<<<dim3(LTOT, NH), 256, 0, stream>>>(Att, attRaw);
    k_msum<<<dim3(2, 38, NH), 32, 0, stream>>>(Att, PMVT, Mb);
    k_proj3<<<dim3(16, 19, 3), 32, 0, stream>>>(Mb, WA, le, le);
  }
  k_cvt_f32_f16<<<(LTOT * HID + 255) / 256, 256, 0, stream>>>(le, le16, LTOT * HID);
  k_aw<<<dim3(2, 19, BB), 256, 0, stream>>>(le16, te16, aw16);
  k_softmax_aw<<<dim3(LTOT, BB), 256, 0, stream>>>(aw16, toks);
  k_feat<<<dim3(1, 19, BB), 256, 0, stream>>>(aw16, teT16, ft16);
  k_final<<<152, 256, 0, stream>>>(ft16, Wout, part);
  k_reduce<<<(BB * LTOT + 255) / 256, 256, 0, stream>>>(part, bout, out);
}
